// TerminatorTagger_46961172415261
// MI455X (gfx1250) — matrix operand order repaired and checked by reading, not compiled, not run on hardware
//
#include <hip/hip_runtime.h>

typedef __attribute__((ext_vector_type(16))) _Float16 v16h;
typedef __attribute__((ext_vector_type(8)))  _Float16 v8h;
typedef __attribute__((ext_vector_type(2)))  __fp16   fp16x2;
typedef __attribute__((ext_vector_type(4)))  __fp16   fp16x4;
typedef __attribute__((ext_vector_type(8)))  __fp16   fp16x8;
typedef __attribute__((ext_vector_type(16))) __fp16   fp16x16;
typedef __attribute__((ext_vector_type(8)))  float    v8f;

constexpr int Bn = 32;
constexpr int Sn = 4096;
constexpr int Hn = 768;
constexpr int Dn = 64;
constexpr int ROWS = Bn * Sn;     // 131072
constexpr int MBLK = 128;         // rows per workgroup (8 waves x 16)
constexpr int KBLK = 32;          // K per WMMA step
constexpr int KP32 = KBLK + 4;    // padded fp32 LDS row: 36 floats = 144B
constexpr int KITERS = Hn / KBLK; // 24
constexpr int W1PACK = KITERS * 4 * 32 * 16;  // 49152 halves in fragment order

__device__ __forceinline__ float relu(float x) {
    // v_med3_f32 x, 0, +inf == clamp to [0, inf) : 1 VALU, no canonicalize pair
    return __builtin_amdgcn_fmed3f(x, 0.f, __builtin_inff());
}

__device__ __forceinline__ v16h cat8(v8h lo, v8h hi) {
    return __builtin_shufflevector(lo, hi, 0,1,2,3,4,5,6,7,8,9,10,11,12,13,14,15);
}

__device__ __forceinline__ fp16x2 pkrelu(float x, float y) {
    return __builtin_amdgcn_cvt_pkrtz(relu(x), relu(y));
}

// ---------------------------------------------------------------------------
// Kernel 0: pack W1 -> f16 in per-lane WMMA B-fragment order (L2-resident).
// ws[((it*4+nt)*32+lane)*16 + i] = (f16) W1[(it*32 + (i&7) + 16*(i>>3) + 8*laneHi) * 64 + nt*16+laneLo]
// ---------------------------------------------------------------------------
__global__ __launch_bounds__(256) void pack_w1_kernel(const float* __restrict__ W1,
                                                      _Float16* __restrict__ wsB) {
    int idx  = blockIdx.x * 256 + threadIdx.x;      // 0 .. W1PACK-1
    int i    = idx & 15;
    int lane = (idx >> 4) & 31;
    int nt   = (idx >> 9) & 3;
    int it   = idx >> 11;
    int k = it * KBLK + (i & 7) + ((i >> 3) << 4) + ((lane >> 4) << 3);
    int n = nt * 16 + (lane & 15);
    wsB[idx] = (_Float16)W1[(size_t)k * Dn + n];
}

// ---------------------------------------------------------------------------
// Issue async copy of one A tile (128x32 fp32) global -> LDS, no VGPR landing.
// Each of the 256 threads issues 4 x b128 (ASYNCcnt-tracked). NT hint: hidden
// is streamed once; keep it from evicting the hot W1/wsB lines in L2.
// ---------------------------------------------------------------------------
__device__ __forceinline__ void issueA(const float* __restrict__ hidden,
                                       int rowBase, int k0, int tid,
                                       float (*__restrict__ lA)[KP32]) {
    #pragma unroll
    for (int t = 0; t < 4; ++t) {
        int idx = tid + t * 256;               // 1024 float4s = 128x32 floats
        int r   = idx >> 3;                    // 8 float4 per row
        int cv  = idx & 7;
        const float* g = hidden + (size_t)(rowBase + r) * Hn + k0 + cv * 4;
        unsigned l = (unsigned)(size_t)&lA[r][cv * 4];   // LDS byte offset (addr[31:0])
        asm volatile("global_load_async_to_lds_b128 %0, %1, off th:TH_LOAD_NT"
                     :: "v"(l), "v"(g) : "memory");
    }
}

// ---------------------------------------------------------------------------
// Kernel 1: fused relu(hidden) @ W1 + b1 -> relu -> @ W2 + b2 -> logits
// ---------------------------------------------------------------------------
__global__ __launch_bounds__(256) void mlp_wmma_kernel(
    const float*    __restrict__ hidden,  // (ROWS, Hn)
    const _Float16* __restrict__ wsB,     // packed W1 fragments (W1PACK halves)
    const float*    __restrict__ b1,      // (Dn)
    const float*    __restrict__ W2,      // (Dn, 2)
    const float*    __restrict__ b2,      // (2)
    float*          __restrict__ out)     // out[0]=loss slot, out+1 = logits
{
    __shared__ __align__(16) float lA32[2][MBLK][KP32];  // 2 x 18 KB fp32 A tiles

    const int tid    = threadIdx.x;
    const int wave   = tid >> 5;          // 0..7
    const int lane   = tid & 31;
    const int laneHi = lane >> 4;         // 0/1 half-wave
    const int laneLo = lane & 15;
    const int rowBase = blockIdx.x * MBLK;
    const int waveRow = wave * 16;        // M offset of this wave inside block

    v8f acc[4] = {};                      // 4 N-tiles of 16 -> N=64

    issueA(hidden, rowBase, 0, tid, lA32[0]);

    for (int it = 0; it < KITERS; ++it) {
        const int cur = it & 1;
        asm volatile("s_wait_asynccnt 0x0" ::: "memory");  // my writes to cur done
        __syncthreads();            // all writes to cur visible; all reads of cur^1 done
        if (it + 1 < KITERS)        // next tile's DMA overlaps this tile's WMMAs
            issueA(hidden, rowBase, (it + 1) * KBLK, tid, lA32[cur ^ 1]);

        // A fragment (ISA 16-bit A 16x32): element i <- lA[m][(i<8 ? i : i+8) + 8*laneHi]
        // read fp32 from LDS, fuse relu + packed f16 convert here
        const float* ar = &lA32[cur][waveRow + laneLo][laneHi * 8];
        float4 c0 = *(const float4*)(ar + 0);
        float4 c1 = *(const float4*)(ar + 4);
        float4 c2 = *(const float4*)(ar + 16);
        float4 c3 = *(const float4*)(ar + 20);
        fp16x2 h0 = pkrelu(c0.x, c0.y), h1 = pkrelu(c0.z, c0.w);
        fp16x2 h2 = pkrelu(c1.x, c1.y), h3 = pkrelu(c1.z, c1.w);
        fp16x2 h4 = pkrelu(c2.x, c2.y), h5 = pkrelu(c2.z, c2.w);
        fp16x2 h6 = pkrelu(c3.x, c3.y), h7 = pkrelu(c3.z, c3.w);
        fp16x4 q0 = __builtin_shufflevector(h0, h1, 0,1,2,3);
        fp16x4 q1 = __builtin_shufflevector(h2, h3, 0,1,2,3);
        fp16x4 q2 = __builtin_shufflevector(h4, h5, 0,1,2,3);
        fp16x4 q3 = __builtin_shufflevector(h6, h7, 0,1,2,3);
        fp16x8 o0 = __builtin_shufflevector(q0, q1, 0,1,2,3,4,5,6,7);
        fp16x8 o1 = __builtin_shufflevector(q2, q3, 0,1,2,3,4,5,6,7);
        fp16x16 aa = __builtin_shufflevector(o0, o1, 0,1,2,3,4,5,6,7,8,9,10,11,12,13,14,15);
        v16h a = __builtin_bit_cast(v16h, aa);

        // B fragments: contiguous 32B per lane from packed-W1 in L2
        const v8h* bp = (const v8h*)(wsB + ((size_t)(it * 4) * 32 + lane) * 16);
        #pragma unroll
        for (int nt = 0; nt < 4; ++nt) {
            v16h bf = cat8(bp[nt * 64 + 0], bp[nt * 64 + 1]);   // nt stride = 512 halves
            acc[nt] = __builtin_amdgcn_wmma_f32_16x16x32_f16(
                false, a, false, bf, (short)0, acc[nt], false, false);
        }
    }

    // ---- fused epilogue: bias+relu then GEMM2 (dot-64 with W2) via shfl_xor
    // lane holds x[m = r + 8*laneHi][n = nt*16 + laneLo] in acc[nt][r]
    float b1v[4], w20[4], w21[4];
    #pragma unroll
    for (int nt = 0; nt < 4; ++nt) {
        int n = nt * 16 + laneLo;
        b1v[nt] = b1[n];
        w20[nt] = W2[n * 2 + 0];
        w21[nt] = W2[n * 2 + 1];
    }
    const float bias2_0 = b2[0], bias2_1 = b2[1];

    #pragma unroll
    for (int r = 0; r < 8; ++r) {
        float p0 = 0.f, p1 = 0.f;
        #pragma unroll
        for (int nt = 0; nt < 4; ++nt) {
            float x = relu(acc[nt][r] + b1v[nt]);
            p0 = fmaf(x, w20[nt], p0);
            p1 = fmaf(x, w21[nt], p1);
        }
        #pragma unroll
        for (int mx = 1; mx <= 8; mx <<= 1) {      // reduce over laneLo (16 lanes)
            p0 += __shfl_xor(p0, mx, 32);
            p1 += __shfl_xor(p1, mx, 32);
        }
        size_t mrow = (size_t)(rowBase + waveRow + r + (laneHi << 3));
        if (laneLo == 0) out[1 + mrow * 2 + 0] = p0 + bias2_0;
        if (laneLo == 1) out[1 + mrow * 2 + 1] = p1 + bias2_1;
    }
}

// ---------------------------------------------------------------------------
// Kernel 2: CRF log-likelihood. One wave32, lane b = batch b.
// ---------------------------------------------------------------------------
__device__ __forceinline__ float lse2(float x, float y) {
    float m = fmaxf(x, y);
    return m + logf(expf(x - m) + expf(y - m));
}

__global__ __launch_bounds__(32) void crf_kernel(
    const float* __restrict__ logits,     // (Bn, Sn, 2) = out+1
    const int*   __restrict__ mask,       // (Bn, Sn)
    const int*   __restrict__ labels,     // (Bn, Sn)
    const float* __restrict__ start_trans,
    const float* __restrict__ end_trans,
    const float* __restrict__ trans,      // (2,2)
    float* __restrict__ out)              // out[0] = loss
{
    __shared__ float red[32];
    const int b = threadIdx.x;
    const float* lg = logits + (size_t)b * Sn * 2;
    const int*   tg = labels + (size_t)b * Sn;
    const int*   mk = mask   + (size_t)b * Sn;

    const float t00 = trans[0], t01 = trans[1], t10 = trans[2], t11 = trans[3];
    const float st0 = start_trans[0], st1 = start_trans[1];
    const float et0 = end_trans[0],   et1 = end_trans[1];

    int   tag0  = tg[0];
    float score = (tag0 ? st1 : st0) + lg[tag0];
    float a0 = st0 + lg[0];
    float a1 = st1 + lg[1];
    int prev = tag0;
    int msum = mk[0];

    for (int t = 1; t < Sn; ++t) {
        int   m  = mk[t];
        float mf = (float)m;
        int   tt = tg[t];
        float e0 = lg[t * 2 + 0];
        float e1 = lg[t * 2 + 1];
        float em = tt ? e1 : e0;
        float tr = prev ? (tt ? t11 : t10) : (tt ? t01 : t00);
        score += (tr + em) * mf;
        float n0 = lse2(a0 + t00, a1 + t10) + e0;
        float n1 = lse2(a0 + t01, a1 + t11) + e1;
        if (m > 0) { a0 = n0; a1 = n1; }
        prev = tt;
        msum += m;
    }

    int last = tg[msum - 1];
    score += last ? et1 : et0;
    float logZ = lse2(a0 + et0, a1 + et1);

    red[b] = score - logZ;
    __syncthreads();
    if (b == 0) {
        float s = 0.f;
        #pragma unroll
        for (int i = 0; i < 32; ++i) s += red[i];
        out[0] = -s;
    }
}

// ---------------------------------------------------------------------------
extern "C" void kernel_launch(void* const* d_in, const int* in_sizes, int n_in,
                              void* d_out, int out_size, void* d_ws, size_t ws_size,
                              hipStream_t stream) {
    const float* hidden = (const float*)d_in[0];
    const int*   mask   = (const int*)  d_in[1];
    const int*   labels = (const int*)  d_in[2];
    const float* W1     = (const float*)d_in[3];
    const float* b1     = (const float*)d_in[4];
    const float* W2     = (const float*)d_in[5];
    const float* b2     = (const float*)d_in[6];
    const float* st     = (const float*)d_in[7];
    const float* et     = (const float*)d_in[8];
    const float* tr     = (const float*)d_in[9];
    float* out = (float*)d_out;
    _Float16* wsB = (_Float16*)d_ws;      // 96 KB of scratch for packed W1

    pack_w1_kernel<<<W1PACK / 256, 256, 0, stream>>>(W1, wsB);
    mlp_wmma_kernel<<<ROWS / MBLK, 256, 0, stream>>>(hidden, wsB, b1, W2, b2, out);
    crf_kernel<<<1, 32, 0, stream>>>(out + 1, mask, labels, st, et, tr, out);
}
